// DenseHistoryCrossBlock_28389733826684
// MI455X (gfx1250) — matrix operand order repaired and checked by reading, not compiled, not run on hardware
//
#include <hip/hip_runtime.h>
#include <hip/hip_bf16.h>
#include <cstdint>

// ---------------------------------------------------------------------------
// CDNA5 (gfx1250) wave32 WMMA bf16 implementation of the DiT block.
// ---------------------------------------------------------------------------

typedef __bf16 bf16_t;
typedef bf16_t v16bf __attribute__((ext_vector_type(16)));
typedef float  v8f   __attribute__((ext_vector_type(8)));

#define WMMA_BF16(a, b, c) \
  __builtin_amdgcn_wmma_f32_16x16x32_bf16(false, (a), false, (b), (short)0, (c), false, false)

static __device__ inline v8f zero8() {
  v8f z;
#pragma unroll
  for (int i = 0; i < 8; ++i) z[i] = 0.f;
  return z;
}

union pk4 { bf16_t h[4]; uint2 u; };

// Async global->LDS 16B DMA (ISA §15.18.3 GLOBAL_LOAD_ASYNC_TO_LDS_B128).
// vdst = 32-bit LDS byte address (low 32 bits of generic LDS pointer),
// vaddr = 64-bit global address, saddr = off (GV mode). Tracked by ASYNCcnt.
static __device__ inline void async_copy_b128(const void* g, void* l) {
  asm volatile("global_load_async_to_lds_b128 %0, %1, off"
               :
               : "v"((uint32_t)(uintptr_t)l), "v"(g)
               : "memory");
}
static __device__ inline void wait_async0() {
#if __has_builtin(__builtin_amdgcn_s_wait_asynccnt)
  __builtin_amdgcn_s_wait_asynccnt(0);
#else
  asm volatile("s_wait_asynccnt 0x0" ::: "memory");
#endif
}

// A-matrix fragment (16x32 bf16) from LDS, row-major rows of length `ld`.
// Per ISA 7.12.2: lanes 0-15 hold rows M=lane with K in {0..7,16..23},
// lanes 16-31 hold rows M=lane-16 with K in {8..15,24..31}; pairs per VGPR.
static __device__ inline v16bf frag_a_lds(const bf16_t* base, int ld, int mBase,
                                          int kBase, int lane) {
  int m = mBase + (lane & 15);
  int koff = (lane < 16) ? 0 : 8;
  v16bf a;
#pragma unroll
  for (int j = 0; j < 8; ++j) {
    int k = kBase + koff + ((j < 4) ? 2 * j : 16 + 2 * (j - 4));
    a[2 * j]     = base[m * ld + k];
    a[2 * j + 1] = base[m * ld + k + 1];
  }
  return a;
}

// B-matrix fragment (32x16 bf16) from LDS stored "transposed": row index = N,
// column index = K (so K pairs are contiguous). lane: N = lane&15,
// lanes 0-15 hold K in {0..7,16..23}, lanes 16-31 hold K in {8..15,24..31}.
static __device__ inline v16bf frag_b_ldsT(const bf16_t* base, int ld, int nBase,
                                           int kBase, int lane) {
  int n = nBase + (lane & 15);
  int kh = (lane >> 4) * 8;
  v16bf b;
#pragma unroll
  for (int j = 0; j < 8; ++j) {
    int k = kBase + kh + ((j < 4) ? 2 * j : 16 + 2 * (j - 4));
    b[2 * j]     = base[n * ld + k];
    b[2 * j + 1] = base[n * ld + k + 1];
  }
  return b;
}

// ---------------------------------------------------------------------------
// bf16 WMMA GEMM with N-major (pre-transposed) weights:
//   C[M,N] = A[M,K] * Wt[N,K]^T (+bias), optional fused X = (X + C) * mask.
// Block = 128 thr (4 waves), tile 64x64, K-step 32. Double-buffered LDS with
// GLOBAL_LOAD_ASYNC_TO_LDS_B128 staging (no VGPR round-trip, no per-chunk
// loadcnt stalls); one s_wait_asynccnt + barrier per K-step.
// ---------------------------------------------------------------------------
__global__ __launch_bounds__(128) void gemm_bf16_kernel(
    const bf16_t* __restrict__ A, const bf16_t* __restrict__ Wt,
    int K, const float* __restrict__ bias,
    float* __restrict__ C, int ldc,
    float* __restrict__ X, const unsigned char* __restrict__ um, int mStart) {
  __shared__ bf16_t As[2][64 * 40];
  __shared__ bf16_t Bs[2][64 * 40];  // [n][k]
  const int tid = threadIdx.x;
  const int lane = tid & 31;
  const int wv = tid >> 5;
  const int mBlk = blockIdx.x * 64;
  const int nBlk = blockIdx.y * 64;

  // tile loader: 64 rows x 32 bf16 per matrix, async 16B DMA into LDS
  auto load_tiles = [&](int k0, int p) {
#pragma unroll
    for (int i = 0; i < 2; ++i) {
      int li = (tid + i * 128) * 8;
      int r = li >> 5, c = li & 31;
      async_copy_b128(A + (size_t)(mBlk + r) * K + k0 + c, &As[p][r * 40 + c]);
      async_copy_b128(Wt + (size_t)(nBlk + r) * K + k0 + c, &Bs[p][r * 40 + c]);
    }
  };

  v8f acc[4];
#pragma unroll
  for (int t = 0; t < 4; ++t) acc[t] = zero8();

  load_tiles(0, 0);
  int p = 0;
  for (int k0 = 0; k0 < K; k0 += 32) {
    wait_async0();        // this wave's async loads into buffer p landed
    __syncthreads();      // all waves landed; reads of buffer p^1 finished
    if (k0 + 32 < K) load_tiles(k0 + 32, p ^ 1);
    v16bf af = frag_a_lds(As[p], 40, wv * 16, 0, lane);
#pragma unroll
    for (int t = 0; t < 4; ++t) {
      v16bf bfg = frag_b_ldsT(Bs[p], 40, t * 16, 0, lane);
      acc[t] = WMMA_BF16(af, bfg, acc[t]);
    }
    p ^= 1;
  }

  const int nLane = lane & 15;
  const int mAdd = (lane >> 4) << 3;
  float bv[4];
#pragma unroll
  for (int t = 0; t < 4; ++t) bv[t] = bias ? bias[nBlk + t * 16 + nLane] : 0.f;

  if (X) {  // residual epilogue: X = (X + v) * mask, row width 512
#pragma unroll
    for (int j = 0; j < 8; ++j) {
      int row = mStart + mBlk + wv * 16 + j + mAdd;
      float mv = um ? (um[(row >> 13) * 128 + (row & 127)] ? 1.f : 0.f) : 1.f;
      float* xr = X + (size_t)row * 512 + nBlk + nLane;
#pragma unroll
      for (int t = 0; t < 4; ++t) {
        float v = acc[t][j] + bv[t];
        xr[t * 16] = (xr[t * 16] + v) * mv;
      }
    }
  } else {
#pragma unroll
    for (int j = 0; j < 8; ++j) {
      float* cr = C + (size_t)(mBlk + wv * 16 + j + mAdd) * ldc + nBlk + nLane;
#pragma unroll
      for (int t = 0; t < 4; ++t) cr[t * 16] = acc[t][j] + bv[t];
    }
  }
}

// ---------------------------------------------------------------------------
// Fused attention (whole KV resident): S = Q K^T * 1/8 (+mask), softmax,
// O = P V.  One block = one (seq, head, 64-row q tile); 4 waves.
// NKT = number of 16-wide key tiles (NKV = 16*NKT).  P aliases the K tile.
// ---------------------------------------------------------------------------
template <int NKT>
__global__ __launch_bounds__(128) void attn_kernel(
    const float* __restrict__ Qp, long long qA, long long qB, long long qC, long long qPos,
    const float* __restrict__ Kp, long long kA, long long kB, long long kC, long long kPos,
    const float* __restrict__ Vp, long long vA, long long vB, long long vC, long long vPos,
    bf16_t* __restrict__ Op, long long oA, long long oB, long long oC, long long oPos,
    int D1, int D2, const unsigned char* __restrict__ um, int maskDiv) {
  constexpr int NKV = NKT * 16;
  constexpr int LDK = 72;
  constexpr int LDV = NKV + 8;
  extern __shared__ char smem_raw[];
  bf16_t* Qs = reinterpret_cast<bf16_t*>(smem_raw);  // 64 x LDK
  bf16_t* Ks = Qs + 64 * LDK;                         // NKV x LDK (aliased by P)
  bf16_t* VsT = Ks + NKV * LDK;                       // 64(d) x LDV(key)
  bf16_t* Ps = Ks;

  const int tid = threadIdx.x;
  const int lane = tid & 31;
  const int wv = tid >> 5;
  const int seq = blockIdx.x;
  const int qt = blockIdx.y;
  const int h = seq % D2;
  const int mid = (seq / D2) % D1;
  const int b = seq / (D2 * D1);
  const long long qBase = (long long)b * qA + (long long)mid * qB + (long long)h * qC;
  const long long kBase = (long long)b * kA + (long long)mid * kB + (long long)h * kC;
  const long long vBase = (long long)b * vA + (long long)mid * vB + (long long)h * vC;

  // Q: float4 loads, packed 4xbf16 (b64) LDS stores
  for (int i = tid; i < 64 * 16; i += 128) {
    int r = i >> 4, d4 = (i & 15) * 4;
    float4 v = *reinterpret_cast<const float4*>(
        &Qp[qBase + (long long)(qt * 64 + r) * qPos + d4]);
    pk4 t;
    t.h[0] = (bf16_t)v.x; t.h[1] = (bf16_t)v.y;
    t.h[2] = (bf16_t)v.z; t.h[3] = (bf16_t)v.w;
    *reinterpret_cast<uint2*>(&Qs[r * LDK + d4]) = t.u;
  }
  // K: same; V: transpose scatter into VsT
  for (int i = tid; i < NKV * 16; i += 128) {
    int r = i >> 4, d4 = (i & 15) * 4;
    float4 kv = *reinterpret_cast<const float4*>(&Kp[kBase + (long long)r * kPos + d4]);
    pk4 t;
    t.h[0] = (bf16_t)kv.x; t.h[1] = (bf16_t)kv.y;
    t.h[2] = (bf16_t)kv.z; t.h[3] = (bf16_t)kv.w;
    *reinterpret_cast<uint2*>(&Ks[r * LDK + d4]) = t.u;
    float4 vv = *reinterpret_cast<const float4*>(&Vp[vBase + (long long)r * vPos + d4]);
    VsT[(d4 + 0) * LDV + r] = (bf16_t)vv.x;
    VsT[(d4 + 1) * LDV + r] = (bf16_t)vv.y;
    VsT[(d4 + 2) * LDV + r] = (bf16_t)vv.z;
    VsT[(d4 + 3) * LDV + r] = (bf16_t)vv.w;
  }
  __syncthreads();

  v8f s[NKT];
#pragma unroll
  for (int t = 0; t < NKT; ++t) s[t] = zero8();
#pragma unroll
  for (int d0 = 0; d0 < 64; d0 += 32) {
    v16bf af = frag_a_lds(Qs, LDK, wv * 16, d0, lane);
#pragma unroll
    for (int t = 0; t < NKT; ++t) {
      v16bf bfg = frag_b_ldsT(Ks, LDK, t * 16, d0, lane);
      s[t] = WMMA_BF16(af, bfg, s[t]);
    }
  }

  const float scale = 0.125f;  // 1/sqrt(64)
  const int nLane = lane & 15;
  const int mAdd = (lane >> 4) << 3;
  if (um) {
    int mBase = (seq / maskDiv) * NKV;
#pragma unroll
    for (int t = 0; t < NKT; ++t) {
      float pen = um[mBase + t * 16 + nLane] ? 0.f : -1e9f;
#pragma unroll
      for (int j = 0; j < 8; ++j) s[t][j] = s[t][j] * scale + pen;
    }
  } else {
#pragma unroll
    for (int t = 0; t < NKT; ++t)
#pragma unroll
      for (int j = 0; j < 8; ++j) s[t][j] *= scale;
  }

  // softmax: (lane,j) holds row m=j+mAdd, col t*16+(lane&15); xor masks 1..8
  // reduce across the 16-lane half that shares a row set.
  float inv[8];
#pragma unroll
  for (int j = 0; j < 8; ++j) {
    float mx = -3.4e38f;
#pragma unroll
    for (int t = 0; t < NKT; ++t) mx = fmaxf(mx, s[t][j]);
#pragma unroll
    for (int msk = 1; msk < 16; msk <<= 1) mx = fmaxf(mx, __shfl_xor(mx, msk, 32));
    float sum = 0.f;
#pragma unroll
    for (int t = 0; t < NKT; ++t) {
      float e = __expf(s[t][j] - mx);
      s[t][j] = e;
      sum += e;
    }
#pragma unroll
    for (int msk = 1; msk < 16; msk <<= 1) sum += __shfl_xor(sum, msk, 32);
    inv[j] = 1.f / sum;
  }

  __syncthreads();  // all waves done reading Ks before P overwrites it
  bf16_t* Pw = Ps + wv * 16 * LDV;
#pragma unroll
  for (int t = 0; t < NKT; ++t)
#pragma unroll
    for (int j = 0; j < 8; ++j)
      Pw[(j + mAdd) * LDV + t * 16 + nLane] = (bf16_t)(s[t][j] * inv[j]);

  v8f o[4];
#pragma unroll
  for (int dt = 0; dt < 4; ++dt) o[dt] = zero8();
#pragma unroll
  for (int ks = 0; ks < NKT / 2; ++ks) {
    v16bf af = frag_a_lds(Pw, LDV, 0, ks * 32, lane);
#pragma unroll
    for (int dt = 0; dt < 4; ++dt) {
      v16bf bfg = frag_b_ldsT(VsT, LDV, dt * 16, ks * 32, lane);
      o[dt] = WMMA_BF16(af, bfg, o[dt]);
    }
  }

  const long long oBase = (long long)b * oA + (long long)mid * oB + (long long)h * oC;
#pragma unroll
  for (int dt = 0; dt < 4; ++dt)
#pragma unroll
    for (int j = 0; j < 8; ++j) {
      int r = qt * 64 + wv * 16 + j + mAdd;
      Op[oBase + (long long)r * oPos + dt * 16 + nLane] = (bf16_t)o[dt][j];
    }
}

// ---------------------------------------------------------------------------
// Fused row kernel: y = LN1(x)*g1+b1 (+mod bias) (*mask), optionally
// z = LN2(y)*g2+b2; writes bf16 row. One block per 512-wide row, 128 threads.
// ---------------------------------------------------------------------------
static __device__ inline float block_sum_512(float v, float* sred) {
#pragma unroll
  for (int m = 16; m >= 1; m >>= 1) v += __shfl_xor(v, m, 32);
  __syncthreads();
  if ((threadIdx.x & 31) == 0) sred[threadIdx.x >> 5] = v;
  __syncthreads();
  return sred[0] + sred[1] + sred[2] + sred[3];
}

__global__ __launch_bounds__(128) void ln_row_kernel(
    const float* __restrict__ in,
    const float* __restrict__ g1, const float* __restrict__ b1,
    const float* __restrict__ modp, int modCol, int rowsPerBatch,
    const unsigned char* __restrict__ um,
    const float* __restrict__ g2, const float* __restrict__ b2,
    bf16_t* __restrict__ out) {
  __shared__ float sred[4];
  const int tid = threadIdx.x;
  const long long row = blockIdx.x;
  const int c0 = tid * 4;
  float4 v4 = reinterpret_cast<const float4*>(in + row * 512)[tid];
  float y[4] = {v4.x, v4.y, v4.z, v4.w};

  float s = y[0] + y[1] + y[2] + y[3];
  s = block_sum_512(s, sred);
  float mean = s * (1.f / 512.f);
  float vs = 0.f;
#pragma unroll
  for (int i = 0; i < 4; ++i) { float d = y[i] - mean; vs += d * d; }
  vs = block_sum_512(vs, sred);
  float rs = rsqrtf(vs * (1.f / 512.f) + 1e-5f);
#pragma unroll
  for (int i = 0; i < 4; ++i) y[i] = (y[i] - mean) * rs * g1[c0 + i] + b1[c0 + i];

  if (modp) {
    int bb = (int)(row / rowsPerBatch);
#pragma unroll
    for (int i = 0; i < 4; ++i) y[i] += modp[bb * 2048 + modCol + c0 + i];
  }
  if (um) {
    float mv = um[(row / rowsPerBatch) * 128 + (row & 127)] ? 1.f : 0.f;
#pragma unroll
    for (int i = 0; i < 4; ++i) y[i] *= mv;
  }
  if (g2) {
    float s2 = y[0] + y[1] + y[2] + y[3];
    s2 = block_sum_512(s2, sred);
    float m2 = s2 * (1.f / 512.f);
    float v2 = 0.f;
#pragma unroll
    for (int i = 0; i < 4; ++i) { float d = y[i] - m2; v2 += d * d; }
    v2 = block_sum_512(v2, sred);
    float rs2 = rsqrtf(v2 * (1.f / 512.f) + 1e-5f);
#pragma unroll
    for (int i = 0; i < 4; ++i) y[i] = (y[i] - m2) * rs2 * g2[c0 + i] + b2[c0 + i];
  }
  pk4 t;
#pragma unroll
  for (int i = 0; i < 4; ++i) t.h[i] = (bf16_t)y[i];
  *reinterpret_cast<uint2*>(out + row * 512 + c0) = t.u;
}

// ---------------------------------------------------------------------------
// Small helpers
// ---------------------------------------------------------------------------
__global__ __launch_bounds__(128) void mod_silu_kernel(
    const float* __restrict__ fc, const float* __restrict__ w,
    const float* __restrict__ bias, float* __restrict__ out) {
  int id = blockIdx.x * 128 + threadIdx.x;  // 4096 = 2*2048
  int b = id >> 11, n = id & 2047;
  float s = bias[n];
  for (int k = 0; k < 512; ++k) {
    float xk = fc[b * 512 + k];
    s += (xk / (1.f + __expf(-xk))) * w[k * 2048 + n];
  }
  out[id] = s;
}

// convert f32 [K,N] weights -> bf16 N-major Wt[n*K + k] (coalesced writes)
__global__ __launch_bounds__(256) void cvt_t_kernel(const float* __restrict__ in,
                                                    bf16_t* __restrict__ out,
                                                    int K, int N) {
  int id = blockIdx.x * 256 + threadIdx.x;
  if (id >= K * N) return;
  int n = id / K, k = id - n * K;
  out[id] = (bf16_t)in[(size_t)k * N + n];
}

// interleaved RoPE, applied in-place per (row, head, pair)
__global__ __launch_bounds__(128) void rope_kernel(
    float* __restrict__ p, long long rowStride, int colBase,
    const float* __restrict__ ang, int dA, int sA, int dB, int mB, long long total) {
  long long id = (long long)blockIdx.x * 128 + threadIdx.x;
  if (id >= total) return;
  int pr = id & 31;
  int hh = (int)((id >> 5) & 7);
  long long row = id >> 8;
  long long aRow = (row / dA) * sA + (row / dB) % mB;
  const float* a = ang + aRow * 64;
  float f0 = a[2 * pr], f1 = a[2 * pr + 1];
  float* q = p + row * rowStride + colBase + hh * 64 + 2 * pr;
  float t0 = q[0], t1 = q[1];
  q[0] = t0 * __cosf(f0) - t1 * __sinf(f0);
  q[1] = t1 * __cosf(f1) + t0 * __sinf(f1);
}

__global__ __launch_bounds__(256) void gelu_gate_kernel(const float* __restrict__ H,
                                                        bf16_t* __restrict__ G, int rows) {
  int id = blockIdx.x * 256 + threadIdx.x;
  if (id >= rows * 2048) return;
  int r = id >> 11, c = id & 2047;
  float hx = H[(size_t)r * 4096 + c];
  float hg = H[(size_t)r * 4096 + 2048 + c];
  float ge = 0.5f * hg * (1.f + erff(hg * 0.70710678118f));
  G[id] = (bf16_t)(hx * ge);
}

// ---------------------------------------------------------------------------
// Host orchestration
// ---------------------------------------------------------------------------
extern "C" void kernel_launch(void* const* d_in, const int* in_sizes, int n_in,
                              void* d_out, int out_size, void* d_ws, size_t ws_size,
                              hipStream_t stream) {
  (void)in_sizes; (void)n_in; (void)out_size;
  constexpr size_t TOK = 16384;           // B*NB*NU
  // workspace layout (bytes)
  constexpr size_t OFF_X    = 0;                         // f32 [16384,512]
  constexpr size_t OFF_A    = OFF_X + TOK * 512 * 4;     // bf16 [16384,512]
  constexpr size_t OFF_CTX  = OFF_A + TOK * 512 * 2;     // bf16 [512,512]
  constexpr size_t OFF_W    = OFF_CTX + 512 * 512 * 2;   // bf16 up to 4096*512
  constexpr size_t OFF_W2   = OFF_W + 512 * 4096 * 2;    // bf16 512*2048
  constexpr size_t OFF_QKV  = OFF_W2 + 2048 * 512 * 2;   // f32 [16384,1536]
  constexpr size_t OFF_H    = OFF_QKV + TOK * 1536 * 4;  // f32 [2048,4096]
  constexpr size_t OFF_MOD  = OFF_H + 2048ull * 4096 * 4;// f32 [2,2048]
  constexpr size_t WS_NEED  = OFF_MOD + 2 * 2048 * 4;
  if (ws_size < WS_NEED) return;

  char* ws = (char*)d_ws;
  float*  X    = (float*)(ws + OFF_X);
  bf16_t* Abuf = (bf16_t*)(ws + OFF_A);
  bf16_t* Cbuf = (bf16_t*)(ws + OFF_CTX);
  bf16_t* Wbf  = (bf16_t*)(ws + OFF_W);
  bf16_t* Wbf2 = (bf16_t*)(ws + OFF_W2);
  float*  QKV  = (float*)(ws + OFF_QKV);
  float*  Hc   = (float*)(ws + OFF_H);
  float*  modb = (float*)(ws + OFF_MOD);
  bf16_t* Gc   = (bf16_t*)QKV;  // QKV region free during FFN

  const float* x_in = (const float*)d_in[0];
  const float* bte  = (const float*)d_in[1];   // [2,64,64]
  const float* enc  = (const float*)d_in[2];   // [2,256,512]
  const float* lte  = (const float*)d_in[3];   // [2,256,64]
  const float* fc   = (const float*)d_in[4];   // [2,512]
  const unsigned char* um = (const unsigned char*)d_in[5];  // [2,128] bool
  // params in make_params() dict order
  const float* p_mod_w   = (const float*)d_in[6];
  const float* p_mod_b   = (const float*)d_in[7];
  const float* p_cn_g    = (const float*)d_in[8];
  const float* p_cn_b    = (const float*)d_in[9];
  const float* p_tn_g    = (const float*)d_in[10];
  const float* p_tn_b    = (const float*)d_in[11];
  const float* p_un_g    = (const float*)d_in[12];
  const float* p_un_b    = (const float*)d_in[13];
  const float* p_fn_g    = (const float*)d_in[14];
  const float* p_fn_b    = (const float*)d_in[15];
  const float* p_ca_ln_g = (const float*)d_in[16];
  const float* p_ca_ln_b = (const float*)d_in[17];
  const float* p_ca_lnc_g= (const float*)d_in[18];
  const float* p_ca_lnc_b= (const float*)d_in[19];
  const float* p_ca_wq   = (const float*)d_in[20];
  const float* p_ca_wkv  = (const float*)d_in[21];
  const float* p_ca_wo   = (const float*)d_in[22];
  const float* p_ca_bo   = (const float*)d_in[23];
  const float* p_ta_ln_g = (const float*)d_in[24];
  const float* p_ta_ln_b = (const float*)d_in[25];
  const float* p_ta_wqkv = (const float*)d_in[26];
  const float* p_ta_wo   = (const float*)d_in[27];
  const float* p_ta_bo   = (const float*)d_in[28];
  const float* p_ua_win  = (const float*)d_in[29];
  const float* p_ua_bin  = (const float*)d_in[30];
  const float* p_ua_wout = (const float*)d_in[31];
  const float* p_ua_bout = (const float*)d_in[32];
  const float* p_ff_ln_g = (const float*)d_in[33];
  const float* p_ff_ln_b = (const float*)d_in[34];
  const float* p_ff_w1   = (const float*)d_in[35];
  const float* p_ff_b1   = (const float*)d_in[36];
  const float* p_ff_w2   = (const float*)d_in[37];
  const float* p_ff_b2   = (const float*)d_in[38];

  auto cvtT = [&](const float* src, bf16_t* dst, int K, int N) {
    cvt_t_kernel<<<(K * N + 255) / 256, 256, 0, stream>>>(src, dst, K, N);
  };

  hipMemcpyAsync(X, x_in, TOK * 512 * 4, hipMemcpyDeviceToDevice, stream);
  mod_silu_kernel<<<32, 128, 0, stream>>>(fc, p_mod_w, p_mod_b, modb);

  // ---------------- cross attention ----------------
  ln_row_kernel<<<16384, 128, 0, stream>>>(X, p_cn_g, p_cn_b, modb, 0, 8192, um,
                                           p_ca_ln_g, p_ca_ln_b, Abuf);
  ln_row_kernel<<<512, 128, 0, stream>>>(enc, p_ca_lnc_g, p_ca_lnc_b, nullptr, 0, 256,
                                         nullptr, nullptr, nullptr, Cbuf);
  cvtT(p_ca_wq, Wbf, 512, 512);
  gemm_bf16_kernel<<<dim3(256, 8), 128, 0, stream>>>(Abuf, Wbf, 512, nullptr,
                                                     QKV, 512, nullptr, nullptr, 0);
  cvtT(p_ca_wkv, Wbf, 512, 1024);
  gemm_bf16_kernel<<<dim3(8, 16), 128, 0, stream>>>(Cbuf, Wbf, 512, nullptr,
                                                    QKV + TOK * 512, 1024, nullptr, nullptr, 0);
  rope_kernel<<<32768, 128, 0, stream>>>(QKV, 512, 0, bte, 8192, 64, 128, 64,
                                         (long long)TOK * 256);
  rope_kernel<<<1024, 128, 0, stream>>>(QKV + TOK * 512, 1024, 0, lte, 256, 256, 1, 256,
                                        (long long)512 * 256);
  {
    size_t sh = (size_t)(64 * 72 + 256 * 72 + 64 * 264) * 2;
    attn_kernel<16><<<dim3(16, 128), 128, sh, stream>>>(
        QKV, 8192ll * 512, 0, 64, 512,
        QKV + TOK * 512, 256ll * 1024, 0, 64, 1024,
        QKV + TOK * 512 + 512, 256ll * 1024, 0, 64, 1024,
        Abuf, 8192ll * 512, 0, 64, 512, 1, 8, nullptr, 1);
  }
  cvtT(p_ca_wo, Wbf, 512, 512);
  gemm_bf16_kernel<<<dim3(256, 8), 128, 0, stream>>>(Abuf, Wbf, 512, p_ca_bo,
                                                     nullptr, 0, X, um, 0);

  // ---------------- temporal attention ----------------
  ln_row_kernel<<<16384, 128, 0, stream>>>(X, p_tn_g, p_tn_b, modb, 512, 8192, nullptr,
                                           p_ta_ln_g, p_ta_ln_b, Abuf);
  cvtT(p_ta_wqkv, Wbf, 512, 1536);
  gemm_bf16_kernel<<<dim3(256, 24), 128, 0, stream>>>(Abuf, Wbf, 512, nullptr,
                                                      QKV, 1536, nullptr, nullptr, 0);
  rope_kernel<<<32768, 128, 0, stream>>>(QKV, 1536, 0, bte, 8192, 64, 128, 64,
                                         (long long)TOK * 256);
  rope_kernel<<<32768, 128, 0, stream>>>(QKV, 1536, 512, bte, 8192, 64, 128, 64,
                                         (long long)TOK * 256);
  {
    size_t sh = (size_t)(64 * 72 + 64 * 72 + 64 * 72) * 2;
    attn_kernel<4><<<dim3(2048, 1), 128, sh, stream>>>(
        QKV,        8192ll * 1536, 1536, 64, 128ll * 1536,
        QKV + 512,  8192ll * 1536, 1536, 64, 128ll * 1536,
        QKV + 1024, 8192ll * 1536, 1536, 64, 128ll * 1536,
        Abuf, 8192ll * 512, 512, 64, 128ll * 512, 128, 8, nullptr, 1);
  }
  cvtT(p_ta_wo, Wbf, 512, 512);
  gemm_bf16_kernel<<<dim3(256, 8), 128, 0, stream>>>(Abuf, Wbf, 512, p_ta_bo,
                                                     nullptr, 0, X, um, 0);

  // ---------------- unit attention ----------------
  ln_row_kernel<<<16384, 128, 0, stream>>>(X, p_un_g, p_un_b, modb, 1024, 8192, nullptr,
                                           nullptr, nullptr, Abuf);
  cvtT(p_ua_win, Wbf, 512, 1536);
  gemm_bf16_kernel<<<dim3(256, 24), 128, 0, stream>>>(Abuf, Wbf, 512, p_ua_bin,
                                                      QKV, 1536, nullptr, nullptr, 0);
  {
    size_t sh = (size_t)(64 * 72 + 128 * 72 + 64 * 136) * 2;
    attn_kernel<8><<<dim3(1024, 2), 128, sh, stream>>>(
        QKV,        8192ll * 1536, 128ll * 1536, 64, 1536,
        QKV + 512,  8192ll * 1536, 128ll * 1536, 64, 1536,
        QKV + 1024, 8192ll * 1536, 128ll * 1536, 64, 1536,
        Abuf, 8192ll * 512, 128ll * 512, 64, 512, 64, 8, um, 512);
  }
  cvtT(p_ua_wout, Wbf, 512, 512);
  gemm_bf16_kernel<<<dim3(256, 8), 128, 0, stream>>>(Abuf, Wbf, 512, p_ua_bout,
                                                     nullptr, 0, X, um, 0);

  // ---------------- FFN (chunked over 8 x 2048 rows) ----------------
  ln_row_kernel<<<16384, 128, 0, stream>>>(X, p_fn_g, p_fn_b, modb, 1536, 8192, nullptr,
                                           p_ff_ln_g, p_ff_ln_b, Abuf);
  cvtT(p_ff_w1, Wbf, 512, 4096);
  cvtT(p_ff_w2, Wbf2, 2048, 512);
  for (int c = 0; c < 8; ++c) {
    const bf16_t* Ac = Abuf + (size_t)c * 2048 * 512;
    gemm_bf16_kernel<<<dim3(32, 64), 128, 0, stream>>>(Ac, Wbf, 512, p_ff_b1,
                                                       Hc, 4096, nullptr, nullptr, 0);
    gelu_gate_kernel<<<16384, 256, 0, stream>>>(Hc, Gc, 2048);
    gemm_bf16_kernel<<<dim3(32, 8), 128, 0, stream>>>(Gc, Wbf2, 2048, p_ff_b2,
                                                      nullptr, 0, X, um, c * 2048);
  }

  hipMemcpyAsync(d_out, X, TOK * 512 * 4, hipMemcpyDeviceToDevice, stream);
}
